// MultiHeadAttention_644245094751
// MI455X (gfx1250) — matrix operand order repaired and checked by reading, not compiled, not run on hardware
//
#include <hip/hip_runtime.h>
#include <hip/hip_bf16.h>

// MI455X / gfx1250 multi-head attention, all GEMMs via v_wmma_f32_16x16x32_bf16.
// bf16-pure staging paths use GLOBAL_LOAD_ASYNC_TO_LDS_B128 (ASYNCcnt) when the
// toolchain exposes the builtins; fp32->bf16 staging uses prefetch + VALU cvt.
// B=4, S=1024, D=1024, H=16, DEPTH=64.

typedef __attribute__((ext_vector_type(8)))  __bf16 v8bf;
typedef __attribute__((ext_vector_type(16))) __bf16 v16bf;
typedef __attribute__((ext_vector_type(8)))  float  v8f;
typedef int v4i __attribute__((vector_size(4 * sizeof(int))));

#define NB   4
#define NS   1024
#define ND   1024
#define NH   16
#define NDEP 64

#if defined(__has_builtin)
#if __has_builtin(__builtin_amdgcn_global_load_async_to_lds_b128) && \
    __has_builtin(__builtin_amdgcn_s_wait_asynccnt)
#define USE_ASYNC_LDS 1
#endif
#endif
#ifndef USE_ASYNC_LDS
#define USE_ASYNC_LDS 0
#endif

// 16-byte global -> LDS copy.  Async path: per-lane global_load_async_to_lds_b128.
__device__ __forceinline__ void copy_b128_to_lds(const __bf16* g, __bf16* l) {
#if USE_ASYNC_LDS
  __builtin_amdgcn_global_load_async_to_lds_b128(
      (v4i*)(uintptr_t)g, (v4i*)(uintptr_t)l, 0, 0);
#else
  *(v8bf*)l = *(const v8bf*)g;
#endif
}

__device__ __forceinline__ void async_copies_done() {
#if USE_ASYNC_LDS
  __builtin_amdgcn_s_wait_asynccnt(0);
#endif
}

__device__ __forceinline__ v8f wmma_bf16(v16bf a, v16bf b, v8f c) {
  // D = A(16x32) * B(32x16) + C, f32 accumulate
  return __builtin_amdgcn_wmma_f32_16x16x32_bf16(
      /*neg_a=*/false, a, /*neg_b=*/false, b,
      /*c_mod=*/(short)0, c, /*reuse_a=*/false, /*reuse_b=*/false);
}

__device__ __forceinline__ v8f zero8() {
  v8f z = {0.f, 0.f, 0.f, 0.f, 0.f, 0.f, 0.f, 0.f};
  return z;
}

// A-matrix fragment (16 x 32 bf16), contiguous k order (same as frag_b):
// lane&15 = M row; lane<16 holds K = 0..15, lane>=16 holds K = 16..31
__device__ __forceinline__ v16bf frag_a(const __bf16* row_base, int ld, int koff, int lane) {
  const __bf16* p = row_base + (lane & 15) * ld + koff + ((lane & 16) ? 16 : 0);
  v8bf lo = *(const v8bf*)(p);
  v8bf hi = *(const v8bf*)(p + 8);
  return __builtin_shufflevector(lo, hi, 0,1,2,3,4,5,6,7,8,9,10,11,12,13,14,15);
}

// B-matrix fragment (32 x 16 bf16): lane&15 = N col; lane<16 holds K=0..15, lane>=16 K=16..31.
// col_base indexes N-major storage (row n contiguous in K).
__device__ __forceinline__ v16bf frag_b(const __bf16* col_base, int ld, int koff, int lane) {
  const __bf16* p = col_base + (lane & 15) * ld + koff + ((lane & 16) ? 16 : 0);
  v8bf lo = *(const v8bf*)(p);
  v8bf hi = *(const v8bf*)(p + 8);
  return __builtin_shufflevector(lo, hi, 0,1,2,3,4,5,6,7,8,9,10,11,12,13,14,15);
}

// ---------------------------------------------------------------------------
// Kernel 1: fused QKV projections.  C[i,j] = sum_k X[i,k]*W[j,k] + b[j]
// grid = (8, 32, 3), block = 256.  Output bf16 in [B,H,S,64] layout.
// ---------------------------------------------------------------------------
__global__ __launch_bounds__(256)
void k_qkv_proj(const float* __restrict__ xq, const float* __restrict__ xk,
                const float* __restrict__ xv,
                const float* __restrict__ wq, const float* __restrict__ bq,
                const float* __restrict__ wk, const float* __restrict__ bk,
                const float* __restrict__ wv, const float* __restrict__ bv,
                __bf16* __restrict__ oq, __bf16* __restrict__ ok,
                __bf16* __restrict__ ov) {
  const float* X; const float* W; const float* bias; __bf16* out;
  if (blockIdx.z == 0)      { X = xq; W = wq; bias = bq; out = oq; }
  else if (blockIdx.z == 1) { X = xk; W = wk; bias = bk; out = ok; }
  else                      { X = xv; W = wv; bias = bv; out = ov; }

  __shared__ __bf16 lA[128 * 40];   // 128 rows x 32 k, pad to 40 (80B row, 16B aligned)
  __shared__ __bf16 lW[128 * 40];

  const int tid  = threadIdx.x;
  const int lane = tid & 31;
  const int wid  = tid >> 5;
  const int wm   = (wid >> 2) * 64;   // wave M offset (2 rows of waves)
  const int wn   = (wid & 3) * 32;    // wave N offset (4 cols of waves)
  const int m0   = blockIdx.y * 128;
  const int n0   = blockIdx.x * 128;

  v8f acc[4][2];
#pragma unroll
  for (int i = 0; i < 4; ++i)
#pragma unroll
    for (int j = 0; j < 2; ++j) acc[i][j] = zero8();

  const int srow = tid >> 1;        // 0..127
  const int scol = (tid & 1) * 16;  // 0 or 16
  const float* gax = X + (size_t)(m0 + srow) * ND + scol;
  const float* gwx = W + (size_t)(n0 + srow) * ND + scol;

  for (int k0 = 0; k0 < ND; k0 += 32) {
    __syncthreads();
    {
      const float* ga = gax + k0;
      __bf16* da = lA + srow * 40 + scol;
#pragma unroll
      for (int e = 0; e < 16; ++e) da[e] = (__bf16)ga[e];
      const float* gw = gwx + k0;
      __bf16* dw = lW + srow * 40 + scol;
#pragma unroll
      for (int e = 0; e < 16; ++e) dw[e] = (__bf16)gw[e];
      if (k0 + 32 < ND) {  // prefetch next k tile (global_prefetch_b8)
        __builtin_prefetch(ga + 32, 0, 0);
        __builtin_prefetch(gw + 32, 0, 0);
      }
    }
    __syncthreads();

    v16bf bf[2];
#pragma unroll
    for (int nt = 0; nt < 2; ++nt)
      bf[nt] = frag_b(lW + (wn + nt * 16) * 40, 40, 0, lane);
#pragma unroll
    for (int mt = 0; mt < 4; ++mt) {
      v16bf af = frag_a(lA + (wm + mt * 16) * 40, 40, 0, lane);
#pragma unroll
      for (int nt = 0; nt < 2; ++nt)
        acc[mt][nt] = wmma_bf16(af, bf[nt], acc[mt][nt]);
    }
  }

  // Epilogue: C/D layout lane&15 = col, vgpr j = row j + 8*(lane>=16).
  const int hsel = (lane & 16) ? 8 : 0;
#pragma unroll
  for (int nt = 0; nt < 2; ++nt) {
    const int gn = n0 + wn + nt * 16 + (lane & 15);
    const float bv_ = bias[gn];
    const int hh = gn >> 6, dd = gn & 63;
#pragma unroll
    for (int mt = 0; mt < 4; ++mt) {
#pragma unroll
      for (int j = 0; j < 8; ++j) {
        const int gm = m0 + wm + mt * 16 + j + hsel;
        const int bb = gm >> 10, ss = gm & 1023;
        out[((size_t)(bb * NH + hh) * NS + ss) * NDEP + dd] =
            (__bf16)(acc[mt][nt][j] + bv_);
      }
    }
  }
}

// ---------------------------------------------------------------------------
// Kernel 2: flash attention per (b,h).  grid = (S/64, B*H), block = 128 (4 waves).
// Each wave owns 16 query rows; scores computed transposed: S^T = K * Q^T.
// K tile staged via async-to-LDS; V staged transposed through VGPRs.
// ---------------------------------------------------------------------------
__global__ __launch_bounds__(128)
void k_attention(const __bf16* __restrict__ Q, const __bf16* __restrict__ K,
                 const __bf16* __restrict__ V,
                 const unsigned char* __restrict__ mask,
                 __bf16* __restrict__ O) {
  __shared__ __bf16 lK[64 * 72];    // keys row-major  [key][d]
  __shared__ __bf16 lVt[64 * 72];   // V transposed    [d][key]

  const int tid  = threadIdx.x;
  const int lane = tid & 31;
  const int w    = tid >> 5;
  const int bh   = blockIdx.y;
  const int b    = bh >> 4;
  const int q0   = blockIdx.x * 64;
  const int hsel = (lane >> 4) & 1;            // lane half
  const int qg   = q0 + w * 16 + (lane & 15);  // this lane's query column

  // Q^T B-fragments (b[e] = Q[q][ds*32 + half*16 + e]), kept in registers.
  const __bf16* qrow = Q + ((size_t)bh * NS + qg) * NDEP;
  v16bf qfrag[2];
#pragma unroll
  for (int dstep = 0; dstep < 2; ++dstep) {
    const __bf16* p = qrow + dstep * 32 + hsel * 16;
    v8bf lo = *(const v8bf*)(p);
    v8bf hi = *(const v8bf*)(p + 8);
    qfrag[dstep] = __builtin_shufflevector(lo, hi, 0,1,2,3,4,5,6,7,8,9,10,11,12,13,14,15);
  }

  v8f o[4];
#pragma unroll
  for (int t = 0; t < 4; ++t) o[t] = zero8();
  float m_run = -__builtin_inff();
  float l_run = 0.f;
  const unsigned char* mrow = mask + ((size_t)b * NS + qg) * NS;

  const int srow = tid >> 1;        // 0..63
  const int scol = (tid & 1) * 32;  // 0 or 32

  for (int kb0 = 0; kb0 < NS; kb0 += 64) {
    __syncthreads();
    {  // stage K tile (async b128 copies) and transposed V tile
      const __bf16* gk = K + ((size_t)bh * NS + kb0 + srow) * NDEP + scol;
      __bf16* dk = lK + srow * 72 + scol;
#pragma unroll
      for (int e = 0; e < 32; e += 8) copy_b128_to_lds(gk + e, dk + e);
      const __bf16* gv = V + ((size_t)bh * NS + kb0 + srow) * NDEP + scol;
#pragma unroll
      for (int e = 0; e < 32; ++e) lVt[(scol + e) * 72 + srow] = gv[e];
      if (kb0 + 64 < NS) {
        __builtin_prefetch(gk + 64 * NDEP, 0, 0);
        __builtin_prefetch(gv + 64 * NDEP, 0, 0);
      }
      async_copies_done();
    }
    __syncthreads();

    // scores^T for 4 tiles of 16 keys; lane holds key rows j + 8*hsel, col = q.
    float sc[4][8];
#pragma unroll
    for (int kt = 0; kt < 4; ++kt) {
      v8f st = zero8();
#pragma unroll
      for (int dstep = 0; dstep < 2; ++dstep) {
        v16bf a = frag_a(lK + kt * 16 * 72, 72, dstep * 32, lane);
        st = wmma_bf16(a, qfrag[dstep], st);
      }
      // mask: 8 consecutive key bytes for this lane half
      const unsigned long long mv =
          *(const unsigned long long*)(mrow + kb0 + kt * 16 + hsel * 8);
#pragma unroll
      for (int j = 0; j < 8; ++j) {
        const float v_ = st[j] * 0.125f;  // 1/sqrt(64)
        sc[kt][j] = ((mv >> (8 * j)) & 0xffull) ? -1e9f : v_;
      }
    }

    // online softmax over this 64-key block (per q column)
    float bm = sc[0][0];
#pragma unroll
    for (int kt = 0; kt < 4; ++kt)
#pragma unroll
      for (int j = 0; j < 8; ++j) bm = fmaxf(bm, sc[kt][j]);
    bm = fmaxf(bm, __shfl_xor(bm, 16, 32));
    const float m_new = fmaxf(m_run, bm);
    const float alpha = __expf(m_run - m_new);
    float p_[4][8];
    float ls = 0.f;
#pragma unroll
    for (int kt = 0; kt < 4; ++kt)
#pragma unroll
      for (int j = 0; j < 8; ++j) {
        p_[kt][j] = __expf(sc[kt][j] - m_new);
        ls += p_[kt][j];
      }
    ls += __shfl_xor(ls, 16, 32);
    l_run = l_run * alpha + ls;
    m_run = m_new;
#pragma unroll
    for (int t = 0; t < 4; ++t) o[t] = o[t] * alpha;

    // O^T += V^T * P^T  (A from lVt; B-fragment of P^T built via half swaps)
#pragma unroll
    for (int kh = 0; kh < 2; ++kh) {
      float t0s[8], t1s[8];
#pragma unroll
      for (int j = 0; j < 8; ++j) {
        t0s[j] = __shfl_xor(p_[kh * 2][j], 16, 32);      // partner rows of tile 2kh
        t1s[j] = __shfl_xor(p_[kh * 2 + 1][j], 16, 32);  // partner rows of tile 2kh+1
      }
      v16bf pb;
#pragma unroll
      for (int e = 0; e < 8; ++e) {
        // half0 lane covers K=0..15 (tile 2kh); half1 covers K=16..31 (tile 2kh+1)
        const float lo = hsel ? t1s[e] : p_[kh * 2][e];
        const float hi = hsel ? p_[kh * 2 + 1][e] : t0s[e];
        pb[e]     = (__bf16)lo;
        pb[8 + e] = (__bf16)hi;
      }
#pragma unroll
      for (int dt = 0; dt < 4; ++dt) {
        v16bf a = frag_a(lVt + dt * 16 * 72, 72, kh * 32, lane);
        o[dt] = wmma_bf16(a, pb, o[dt]);
      }
    }
  }

  // normalize and store: O^T tile rows are d = dt*16 + j + 8*hsel, col = q.
  const float inv_l = (l_run > 0.f) ? (1.f / l_run) : 0.f;
  __bf16* orow = O + ((size_t)bh * NS + qg) * NDEP;
#pragma unroll
  for (int dt = 0; dt < 4; ++dt) {
    v8bf pk;
#pragma unroll
    for (int j = 0; j < 8; ++j) pk[j] = (__bf16)(o[dt][j] * inv_l);
    *(v8bf*)(orow + dt * 16 + hsel * 8) = pk;
  }
}

// ---------------------------------------------------------------------------
// Kernel 3: output FC.  out[i,j] = sum_k A[i,k]*fc_w[j,k] + fc_b[j], f32 out.
// A gathered from head-major bf16 workspace via async-to-LDS b128 copies.
// grid = (8, 32), block = 256.
// ---------------------------------------------------------------------------
__global__ __launch_bounds__(256)
void k_fc(const __bf16* __restrict__ A, const float* __restrict__ W,
          const float* __restrict__ bias, float* __restrict__ out) {
  __shared__ __bf16 lA[128 * 40];
  __shared__ __bf16 lW[128 * 40];

  const int tid  = threadIdx.x;
  const int lane = tid & 31;
  const int wid  = tid >> 5;
  const int wm   = (wid >> 2) * 64;
  const int wn   = (wid & 3) * 32;
  const int m0   = blockIdx.y * 128;
  const int n0   = blockIdx.x * 128;

  v8f acc[4][2];
#pragma unroll
  for (int i = 0; i < 4; ++i)
#pragma unroll
    for (int j = 0; j < 2; ++j) acc[i][j] = zero8();

  const int srow = tid >> 1;
  const int scol = (tid & 1) * 16;
  const int gm_s = m0 + srow;
  const int bb_s = gm_s >> 10, ss_s = gm_s & 1023;
  const float* gwx = W + (size_t)(n0 + srow) * ND + scol;

  for (int k0 = 0; k0 < ND; k0 += 32) {
    __syncthreads();
    {
      const int hh = k0 >> 6, off = (k0 & 63) + scol;  // 16-elem chunk stays in one head
      const __bf16* ga = A + ((size_t)(bb_s * NH + hh) * NS + ss_s) * NDEP + off;
      __bf16* da = lA + srow * 40 + scol;
      copy_b128_to_lds(ga, da);
      copy_b128_to_lds(ga + 8, da + 8);
      const float* gw = gwx + k0;
      __bf16* dw = lW + srow * 40 + scol;
#pragma unroll
      for (int e = 0; e < 16; ++e) dw[e] = (__bf16)gw[e];
      if (k0 + 32 < ND) __builtin_prefetch(gw + 32, 0, 0);
      async_copies_done();
    }
    __syncthreads();

    v16bf bf[2];
#pragma unroll
    for (int nt = 0; nt < 2; ++nt)
      bf[nt] = frag_b(lW + (wn + nt * 16) * 40, 40, 0, lane);
#pragma unroll
    for (int mt = 0; mt < 4; ++mt) {
      v16bf af = frag_a(lA + (wm + mt * 16) * 40, 40, 0, lane);
#pragma unroll
      for (int nt = 0; nt < 2; ++nt)
        acc[mt][nt] = wmma_bf16(af, bf[nt], acc[mt][nt]);
    }
  }

  const int hsel = (lane & 16) ? 8 : 0;
#pragma unroll
  for (int nt = 0; nt < 2; ++nt) {
    const int gn = n0 + wn + nt * 16 + (lane & 15);
    const float bv_ = bias[gn];
#pragma unroll
    for (int mt = 0; mt < 4; ++mt) {
#pragma unroll
      for (int j = 0; j < 8; ++j) {
        const int gm = m0 + wm + mt * 16 + j + hsel;
        out[(size_t)gm * ND + gn] = acc[mt][nt][j] + bv_;
      }
    }
  }
}

// ---------------------------------------------------------------------------
extern "C" void kernel_launch(void* const* d_in, const int* in_sizes, int n_in,
                              void* d_out, int out_size, void* d_ws, size_t ws_size,
                              hipStream_t stream) {
  const float* query = (const float*)d_in[0];
  const float* key   = (const float*)d_in[1];
  const float* value = (const float*)d_in[2];
  // jnp bool array: 1 byte per element
  const unsigned char* mask = (const unsigned char*)d_in[3];
  const float* wq_w = (const float*)d_in[4];
  const float* wq_b = (const float*)d_in[5];
  const float* wk_w = (const float*)d_in[6];
  const float* wk_b = (const float*)d_in[7];
  const float* wv_w = (const float*)d_in[8];
  const float* wv_b = (const float*)d_in[9];
  const float* fc_w = (const float*)d_in[10];
  const float* fc_b = (const float*)d_in[11];
  float* out = (float*)d_out;

  const size_t n = (size_t)NB * NS * ND;  // 4M elements
  __bf16* qws = (__bf16*)d_ws;            // 8 MB each, 32 MB total
  __bf16* kws = qws + n;
  __bf16* vws = kws + n;
  __bf16* aws = vws + n;

  k_qkv_proj<<<dim3(8, 32, 3), 256, 0, stream>>>(
      query, key, value, wq_w, wq_b, wk_w, wk_b, wv_w, wv_b, qws, kws, vws);
  k_attention<<<dim3(NS / 64, NB * NH), 128, 0, stream>>>(
      qws, kws, vws, mask, aws);
  k_fc<<<dim3(8, 32), 256, 0, stream>>>(aws, fc_w, fc_b, out);
}
